// MultiHeadAttention_64278480552082
// MI455X (gfx1250) — matrix operand order repaired and checked by reading, not compiled, not run on hardware
//
#include <hip/hip_runtime.h>

typedef __attribute__((ext_vector_type(16))) _Float16 v16h;
typedef __attribute__((ext_vector_type(8)))  float    v8f;

#define BB 4
#define CC 512
#define LL 2048
#define HH 8
#define DD 64

// ---------- WMMA wrapper: D = A(16x32 f16) * B(32x16 f16) + C(16x16 f32) ----------
static __device__ inline v8f wmma_f16(v16h a, v16h b, v8f c) {
  return __builtin_amdgcn_wmma_f32_16x16x32_f16(false, a, false, b, (short)0, c,
                                                false, false);
}

// A fragment: logical [m][k], storage row-major [row=m][k contiguous], stride in dwords.
// Layout (ISA 7.12.2): lane m=lane&15; VGPR v: k = (v>>2)*16 + (lane>>4)*8 + (v&3)*2.
static __device__ inline v16h load_fragA(const unsigned* p, int row, int sdw) {
  const int lane = threadIdx.x & 31;
  const int hi = lane >> 4;
  union { v16h h; unsigned u[8]; } f;
  const unsigned* r = p + row * sdw;
#pragma unroll
  for (int v = 0; v < 8; ++v) {
    int off = ((v >> 2) << 3) + (hi << 2) + (v & 3);  // dword index of k-pair
    f.u[v] = r[off];
  }
  return f.h;
}

// B fragment: logical [k][n], storage holds transpose: [row=n][k contiguous].
// Layout: lane n=lane&15; VGPR v: k = (v>>2)*16 + (lane>>4)*8 + (v&3)*2.
static __device__ inline v16h load_fragB(const unsigned* p, int row, int sdw) {
  const int lane = threadIdx.x & 31;
  const int hi = lane >> 4;
  union { v16h h; unsigned u[8]; } f;
  const unsigned* r = p + row * sdw;
#pragma unroll
  for (int v = 0; v < 8; ++v) f.u[v] = r[((v >> 2) << 3) + (hi << 2) + (v & 3)];
  return f.h;
}

// ---------- prep kernels ----------
// x[B,C,L] f32 -> Xt[B,L,C] f16
__global__ __launch_bounds__(256) void k_cvt_transpose(const float* __restrict__ x,
                                                       _Float16* __restrict__ xt) {
  int idx = blockIdx.x * 256 + threadIdx.x;          // b*L*C + l*C + c
  int c = idx & (CC - 1);
  int rest = idx >> 9;
  int l = rest & (LL - 1);
  int b = rest >> 11;
  xt[idx] = (_Float16)x[((b << 9) + c) * LL + l];
}

// w[C,C,3] f32 -> Wp[t][co][ci] f16
__global__ __launch_bounds__(256) void k_pack_w(const float* __restrict__ w,
                                                _Float16* __restrict__ wp) {
  int idx = blockIdx.x * 256 + threadIdx.x;          // t*C*C + co*C + ci
  int ci = idx & (CC - 1);
  int rest = idx >> 9;
  int co = rest & (CC - 1);
  int t = rest >> 9;
  wp[idx] = (_Float16)w[(co * CC + ci) * 3 + t];
}

// ---------- conv1d(k=3,'same') as WMMA GEMM, double-buffered LDS ----------
// Y[b,co,l] = bias[co] + sum_t sum_ci Wp[t][co][ci] * Xt[b][l+t-1][ci]
// grid: (L/128, C/64, B), block 256 (8 waves). Block tile 64co x 128l;
// wave w computes 32x32: co sub = (w&1)*32, l sub = (w>>1)*32.
// K loop: 24 rounds of 64 (3 taps x 8 ci-chunks), ONE barrier per round.
#define KCH   64                 // K staged per round
#define LDSTR 72                 // row stride in halfs (36 dwords): conflict-free, 16B-aligned
#define ASZ   (64 * LDSTR)       // A buffer halfs
#define BSZ   (128 * LDSTR)      // B buffer halfs

__global__ __launch_bounds__(256) void k_conv_gemm(const _Float16* __restrict__ Wp,
                                                   const float* __restrict__ bias,
                                                   const _Float16* __restrict__ Xt,
                                                   _Float16* __restrict__ outH,
                                                   float* __restrict__ outF,
                                                   int f16out) {
  __shared__ alignas(16) _Float16 lds[2][ASZ + BSZ];   // 54 KB
  const int tid = threadIdx.x;
  const int lane = tid & 31, w = tid >> 5;
  const int hi = lane >> 4, ln = lane & 15;
  const int l0 = blockIdx.x * 128, co0 = blockIdx.y * 64, b = blockIdx.z;
  const int coW = (w & 1) * 32, lW = (w >> 1) * 32;
  const int srow = tid >> 3, scc = tid & 7;            // stager: row=tid/8, 16B chunk=tid%8

  v8f acc00, acc01, acc10, acc11;
#pragma unroll
  for (int e = 0; e < 8; ++e) { acc00[e] = 0.f; acc01[e] = 0.f; acc10[e] = 0.f; acc11[e] = 0.f; }

  // ---- prologue: stage round 0 into buffer 0 ----
  {
    const int t = 0, ci0 = 0;
#pragma unroll
    for (int i = 0; i < 2; ++i) {   // A: 64 rows x 8 chunks
      int row = srow + i * 32;
      float4 av = ((const float4*)Wp)[(t * CC * CC + (co0 + row) * CC + ci0 + scc * 8) >> 3];
      *(float4*)(&lds[0][row * LDSTR + scc * 8]) = av;
    }
#pragma unroll
    for (int i = 0; i < 4; ++i) {   // B: 128 rows x 8 chunks
      int row = srow + i * 32;
      int ls = l0 + row + t - 1;
      float4 bv = make_float4(0.f, 0.f, 0.f, 0.f);
      if (ls >= 0 && ls < LL)
        bv = ((const float4*)Xt)[(b * LL * CC + ls * CC + ci0 + scc * 8) >> 3];
      *(float4*)(&lds[0][ASZ + row * LDSTR + scc * 8]) = bv;
    }
  }

  for (int r = 0; r < 24; ++r) {
    __syncthreads();   // buf[r&1] staged by all waves; buf[(r+1)&1] fully consumed

    // ---- issue global loads for round r+1 (overlap with WMMAs below) ----
    float4 ga[2], gb[4];
    if (r < 23) {
      const int t2 = (r + 1) >> 3, ci2 = ((r + 1) & 7) * KCH;
#pragma unroll
      for (int i = 0; i < 2; ++i) {
        int row = srow + i * 32;
        ga[i] = ((const float4*)Wp)[(t2 * CC * CC + (co0 + row) * CC + ci2 + scc * 8) >> 3];
      }
#pragma unroll
      for (int i = 0; i < 4; ++i) {
        int row = srow + i * 32;
        int ls = l0 + row + t2 - 1;
        gb[i] = make_float4(0.f, 0.f, 0.f, 0.f);
        if (ls >= 0 && ls < LL)
          gb[i] = ((const float4*)Xt)[(b * LL * CC + ls * CC + ci2 + scc * 8) >> 3];
      }
    }

    // ---- compute: 2 k-chunks of 32, 4 WMMAs each ----
    const unsigned* AU = (const unsigned*)&lds[r & 1][0];
    const unsigned* BU = (const unsigned*)&lds[r & 1][ASZ];
#pragma unroll
    for (int ch = 0; ch < 2; ++ch) {
      v16h a0 = load_fragA(AU + ch * 16, coW + ln, LDSTR / 2);
      v16h a1 = load_fragA(AU + ch * 16, coW + 16 + ln, LDSTR / 2);
      v16h b0 = load_fragB(BU + ch * 16, lW + ln, LDSTR / 2);
      v16h b1 = load_fragB(BU + ch * 16, lW + 16 + ln, LDSTR / 2);
      acc00 = wmma_f16(a0, b0, acc00);
      acc01 = wmma_f16(a0, b1, acc01);
      acc10 = wmma_f16(a1, b0, acc10);
      acc11 = wmma_f16(a1, b1, acc11);
    }

    // ---- store staged regs into the other buffer ----
    if (r < 23) {
      _Float16* An = &lds[(r + 1) & 1][0];
      _Float16* Bn = &lds[(r + 1) & 1][ASZ];
#pragma unroll
      for (int i = 0; i < 2; ++i)
        *(float4*)(&An[(srow + i * 32) * LDSTR + scc * 8]) = ga[i];
#pragma unroll
      for (int i = 0; i < 4; ++i)
        *(float4*)(&Bn[(srow + i * 32) * LDSTR + scc * 8]) = gb[i];
    }
  }

  // ---- epilogue: C-layout element (m = v + 8*hi, n = lane&15) ----
#pragma unroll
  for (int a = 0; a < 2; ++a) {
    v8f* accs = (a == 0) ? &acc00 : &acc10;   // accs[0]=l-tile0, accs[1]=l-tile1
#pragma unroll
    for (int v = 0; v < 8; ++v) {
      int co = co0 + coW + a * 16 + v + 8 * hi;
      float bb = bias[co];
      int l = l0 + lW + ln;
      float o0 = accs[0][v] + bb;
      float o1 = accs[1][v] + bb;
      int base = b * CC * LL + co * LL;
      if (f16out) {
        outH[base + l]      = (_Float16)o0;
        outH[base + l + 16] = (_Float16)o1;
      } else {
        outF[base + l]      = o0;
        outF[base + l + 16] = o1;
      }
    }
  }
}

// ---------- flash attention ----------
// Q/K/V f16 flat [b][c*L+l]; attention view row (l',h): 64 contiguous halfs at l'*512+h*64.
// grid: (L/128, H, B), block 256 (8 waves x 16 q-rows). Output written directly into
// transposed layout Oxt[b][lc][c] consumed by the final conv.
__global__ __launch_bounds__(256) void k_flash_attn(const _Float16* __restrict__ Qf,
                                                    const _Float16* __restrict__ Kf,
                                                    const _Float16* __restrict__ Vf,
                                                    _Float16* __restrict__ Oxt) {
  __shared__ alignas(16) _Float16 ldsK[64 * 72];    // [kv row][d]
  __shared__ alignas(16) _Float16 ldsVt[64 * 72];   // [d][kv row]
  __shared__ alignas(16) _Float16 ldsP[128 * 72];   // [q row local][kv]
  const int tid = threadIdx.x;
  const int w = tid >> 5, lane = tid & 31;
  const int hi = lane >> 4, ln = lane & 15;
  const int h = blockIdx.y, b = blockIdx.z;
  const int q0 = blockIdx.x * 128 + w * 16;
  const int CL = CC * LL;

  // Q fragments held in registers for the whole kernel (rows contiguous in d)
  const unsigned* Qu = (const unsigned*)Qf + (b * CL >> 1) + q0 * (CC >> 1) + h * (DD >> 1);
  v16h aQ0 = load_fragA(Qu, ln, 256);
  v16h aQ1 = load_fragA(Qu + 16, ln, 256);

  float rmax[8], rsum[8];
  v8f ot[4];
#pragma unroll
  for (int v = 0; v < 8; ++v) { rmax[v] = -3.0e38f; rsum[v] = 0.f; }
#pragma unroll
  for (int j = 0; j < 4; ++j)
#pragma unroll
    for (int e = 0; e < 8; ++e) ot[j][e] = 0.f;

  const unsigned* KU = (const unsigned*)ldsK;
  const unsigned* VU = (const unsigned*)ldsVt;
  const unsigned* PU = (const unsigned*)ldsP + w * 16 * 36;

  for (int kv0 = 0; kv0 < LL; kv0 += 64) {
    __syncthreads();
#pragma unroll
    for (int i = 0; i < 2; ++i) {
      int cidx = tid + i * 256;
      int r = cidx >> 3, cc = cidx & 7;
      int gi = (b * CL + (kv0 + r) * CC + h * DD + cc * 8) >> 3;
      float4 k4 = ((const float4*)Kf)[gi];
      *(float4*)(ldsK + r * 72 + cc * 8) = k4;
      float4 v4 = ((const float4*)Vf)[gi];
      union { float4 f; _Float16 hh[8]; } uu;
      uu.f = v4;
#pragma unroll
      for (int e = 0; e < 8; ++e) ldsVt[(cc * 8 + e) * 72 + r] = uu.hh[e];
    }
    __syncthreads();

    // S = Q K^T  (16 q-rows x 64 kv-cols per wave)
    v8f st[4];
#pragma unroll
    for (int j = 0; j < 4; ++j)
#pragma unroll
      for (int e = 0; e < 8; ++e) st[j][e] = 0.f;
#pragma unroll
    for (int j = 0; j < 4; ++j) {
      v16h bk = load_fragB(KU, j * 16 + ln, 36);
      st[j] = wmma_f16(aQ0, bk, st[j]);
    }
#pragma unroll
    for (int j = 0; j < 4; ++j) {
      v16h bk = load_fragB(KU + 16, j * 16 + ln, 36);
      st[j] = wmma_f16(aQ1, bk, st[j]);
    }

    // scale by 1/D (reference divides by attention_size, not sqrt)
    const float sc = 1.0f / 64.0f;
    float tmax[8];
#pragma unroll
    for (int v = 0; v < 8; ++v) {
      float m = -3.0e38f;
#pragma unroll
      for (int j = 0; j < 4; ++j) {
        st[j][v] *= sc;
        m = fmaxf(m, st[j][v]);
      }
      tmax[v] = m;
    }
    // row max across the 16 lanes holding each row (xor stays within half)
#pragma unroll
    for (int m = 1; m <= 8; m <<= 1)
#pragma unroll
      for (int v = 0; v < 8; ++v) tmax[v] = fmaxf(tmax[v], __shfl_xor(tmax[v], m, 32));

    float corr[8], psum[8];
#pragma unroll
    for (int v = 0; v < 8; ++v) {
      float nm = fmaxf(rmax[v], tmax[v]);
      corr[v] = __expf(rmax[v] - nm);
      rmax[v] = nm;
      psum[v] = 0.f;
    }
#pragma unroll
    for (int j = 0; j < 4; ++j)
#pragma unroll
      for (int v = 0; v < 8; ++v) {
        float p = __expf(st[j][v] - rmax[v]);
        st[j][v] = p;
        psum[v] += p;
      }
#pragma unroll
    for (int m = 1; m <= 8; m <<= 1)
#pragma unroll
      for (int v = 0; v < 8; ++v) psum[v] += __shfl_xor(psum[v], m, 32);
#pragma unroll
    for (int v = 0; v < 8; ++v) rsum[v] = rsum[v] * corr[v] + psum[v];
#pragma unroll
    for (int j = 0; j < 4; ++j)
#pragma unroll
      for (int v = 0; v < 8; ++v) ot[j][v] *= corr[v];

    // P: C-layout -> A-layout via per-wave LDS region
#pragma unroll
    for (int j = 0; j < 4; ++j)
#pragma unroll
      for (int v = 0; v < 8; ++v)
        ldsP[(w * 16 + v + 8 * hi) * 72 + j * 16 + ln] = (_Float16)st[j][v];

    // O += P @ V
#pragma unroll
    for (int ch = 0; ch < 2; ++ch) {
      v16h aP = load_fragA(PU + ch * 16, ln, 36);
#pragma unroll
      for (int j = 0; j < 4; ++j) {
        v16h bv = load_fragB(VU + ch * 16, j * 16 + ln, 36);
        ot[j] = wmma_f16(aP, bv, ot[j]);
      }
    }
  }

  // normalize + scatter into transposed layout for the final conv:
  // flat f = l'*512 + h*64 + d == c*2048 + lc  ->  c = l'>>2, lc = (l'&3)*512 + h*64 + d
  float inv[8];
#pragma unroll
  for (int v = 0; v < 8; ++v) inv[v] = 1.0f / rsum[v];
#pragma unroll
  for (int j = 0; j < 4; ++j)
#pragma unroll
    for (int v = 0; v < 8; ++v) {
      int lq = q0 + v + 8 * hi;
      int d = j * 16 + ln;
      int c = lq >> 2;
      int lc = ((lq & 3) << 9) + h * DD + d;
      Oxt[b * CL + lc * CC + c] = (_Float16)(ot[j][v] * inv[v]);
    }
}

// ---------- host ----------
extern "C" void kernel_launch(void* const* d_in, const int* in_sizes, int n_in,
                              void* d_out, int out_size, void* d_ws, size_t ws_size,
                              hipStream_t stream) {
  (void)in_sizes; (void)n_in; (void)out_size; (void)ws_size;
  const float* q    = (const float*)d_in[0];
  const float* k    = (const float*)d_in[1];
  const float* v    = (const float*)d_in[2];
  const float* wq_w = (const float*)d_in[3];
  const float* wq_b = (const float*)d_in[4];
  const float* wk_w = (const float*)d_in[5];
  const float* wk_b = (const float*)d_in[6];
  const float* wv_w = (const float*)d_in[7];
  const float* wv_b = (const float*)d_in[8];
  const float* fc_w = (const float*)d_in[9];
  const float* fc_b = (const float*)d_in[10];
  float* out = (float*)d_out;

  const size_t SZ_XT = (size_t)BB * LL * CC * sizeof(_Float16);  // 8 MiB
  const size_t SZ_WP = (size_t)3 * CC * CC * sizeof(_Float16);   // 1.5 MiB
  char* p = (char*)d_ws;
  _Float16* XtQ = (_Float16*)(p);
  _Float16* XtK = (_Float16*)(p + SZ_XT);
  _Float16* XtV = (_Float16*)(p + 2 * SZ_XT);
  _Float16* WpQ = (_Float16*)(p + 3 * SZ_XT);
  _Float16* WpK = (_Float16*)(p + 3 * SZ_XT + SZ_WP);
  _Float16* WpV = (_Float16*)(p + 3 * SZ_XT + 2 * SZ_WP);
  _Float16* WpF = (_Float16*)(p + 3 * SZ_XT + 3 * SZ_WP);
  char* p2 = p + 3 * SZ_XT + 4 * SZ_WP;
  _Float16* Qf  = (_Float16*)(p2);
  _Float16* Kf  = (_Float16*)(p2 + SZ_XT);
  _Float16* Vf  = (_Float16*)(p2 + 2 * SZ_XT);
  _Float16* Axt = (_Float16*)(p2 + 3 * SZ_XT);

  const int nCvt = (BB * LL * CC) / 256;   // 16384
  const int nPk  = (3 * CC * CC) / 256;    // 3072
  k_cvt_transpose<<<nCvt, 256, 0, stream>>>(q, XtQ);
  k_cvt_transpose<<<nCvt, 256, 0, stream>>>(k, XtK);
  k_cvt_transpose<<<nCvt, 256, 0, stream>>>(v, XtV);
  k_pack_w<<<nPk, 256, 0, stream>>>(wq_w, WpQ);
  k_pack_w<<<nPk, 256, 0, stream>>>(wk_w, WpK);
  k_pack_w<<<nPk, 256, 0, stream>>>(wv_w, WpV);
  k_pack_w<<<nPk, 256, 0, stream>>>(fc_w, WpF);

  dim3 gConv(LL / 128, CC / 64, BB);       // (16, 8, 4)
  k_conv_gemm<<<gConv, 256, 0, stream>>>(WpQ, wq_b, XtQ, Qf, nullptr, 1);
  k_conv_gemm<<<gConv, 256, 0, stream>>>(WpK, wk_b, XtK, Kf, nullptr, 1);
  k_conv_gemm<<<gConv, 256, 0, stream>>>(WpV, wv_b, XtV, Vf, nullptr, 1);

  dim3 gAttn(LL / 128, HH, BB);            // (16, 8, 4)
  k_flash_attn<<<gAttn, 256, 0, stream>>>(Qf, Kf, Vf, Axt);

  k_conv_gemm<<<gConv, 256, 0, stream>>>(WpF, fc_b, Axt, nullptr, out, 0);
}
